// PredGAT_87797721465382
// MI455X (gfx1250) — matrix operand order repaired and checked by reading, not compiled, not run on hardware
//
#include <hip/hip_runtime.h>

// MI455X / gfx1250, wave32. All large matmuls via v_wmma_f32_16x16x32_bf16.
// Workspace: 23552*(64+5*128+2*256)+512 f32  +  ~0.57MB bf16 weights
//            + 6MB bf16 K + 6MB bf16 V^T  ~= 127 MB.

typedef __attribute__((ext_vector_type(16))) __bf16 v16bf;
typedef __attribute__((ext_vector_type(8)))  float  v8f;
typedef __attribute__((ext_vector_type(2)))  float  v2f;

#define NNODE 23
#define BATCH 1024
#define ROWS  (BATCH * NNODE)   // 23552, multiple of 16
#define EPSBN 1e-5f

// ---------------------------------------------------------------------------
// BN stats: per node channel n (23 channels), mean/var over (batch, feat).
// ---------------------------------------------------------------------------
__global__ void bn_stats_kernel(const float* __restrict__ in, int F,
                                float* __restrict__ meanOut,
                                float* __restrict__ rstdOut) {
  int n   = blockIdx.x;
  int tid = threadIdx.x;
  int total = BATCH * F;
  float sum = 0.f, sumsq = 0.f;
  for (int idx = tid; idx < total; idx += blockDim.x) {
    int b = idx / F, f = idx % F;
    float v = in[((long)b * NNODE + n) * F + f];
    sum += v;  sumsq += v * v;
  }
  __shared__ float ssum[256], ssq[256];
  ssum[tid] = sum; ssq[tid] = sumsq;
  __syncthreads();
  for (int s = 128; s > 0; s >>= 1) {
    if (tid < s) { ssum[tid] += ssum[tid + s]; ssq[tid] += ssq[tid + s]; }
    __syncthreads();
  }
  if (tid == 0) {
    float m = ssum[0] / (float)total;
    float var = ssq[0] / (float)total - m * m;
    meanOut[n] = m;
    rstdOut[n] = rsqrtf(var + EPSBN);
  }
}

// ---------------------------------------------------------------------------
// fp32 -> bf16 elementwise conversion.
// ---------------------------------------------------------------------------
__global__ void cvt_bf16_kernel(const float* __restrict__ src,
                                __bf16* __restrict__ dst, int n) {
  int i = blockIdx.x * blockDim.x + threadIdx.x;
  if (i < n) dst[i] = (__bf16)src[i];
}

// ---------------------------------------------------------------------------
// V pack+transpose: vp [B,23,128] f32 (cols h*32+d)  ->  vt [n][h][d][m] bf16.
// ---------------------------------------------------------------------------
__global__ void v_pack_kernel(const float* __restrict__ vp,
                              __bf16* __restrict__ vt) {
  int idx = blockIdx.x * blockDim.x + threadIdx.x;  // NNODE*4*32*BATCH
  if (idx >= NNODE * 4 * 32 * BATCH) return;
  int m    = idx & (BATCH - 1);
  int rest = idx >> 10;         // (n*4+h)*32 + d
  int d    = rest & 31;
  int nh   = rest >> 5;
  int h = nh & 3, n = nh >> 2;
  vt[idx] = (__bf16)vp[((long)m * NNODE + n) * 128 + h * 32 + d];
}

// ---------------------------------------------------------------------------
// Guard-free fused GEMM: out = act( (in*s + t) @ Wb^T + bias ).
// Requirements: M%16==0, K%32==0, Nout%64==0.
// in: [M,K] f32; Wb: [Nout,K] bf16; out: [M,Nout] f32.
// BN affine per row: s = gamma[n]*rstd[n], t = beta[n]-mean[n]*s, n = row%23.
// One wave computes a 16x64 tile: one A fragment feeds 4 WMMAs per K-step.
// act: 0=none, 1=leakyrelu(0.2), 2=relu.
// ---------------------------------------------------------------------------
__global__ void gemm_bn_wmma_kernel(const float* __restrict__ in,
                                    const __bf16* __restrict__ Wb,
                                    const float* __restrict__ bias,
                                    const float* __restrict__ mean,
                                    const float* __restrict__ rstd,
                                    const float* __restrict__ gamma,
                                    const float* __restrict__ beta,
                                    float* __restrict__ out,
                                    int M, int K, int Nout, int act, int useBN) {
  int lane = threadIdx.x & 31;
  int wave = threadIdx.x >> 5;
  int ngroups = Nout >> 6;            // groups of 64 output cols
  int mtiles  = M >> 4;
  int tile = blockIdx.x * (blockDim.x >> 5) + wave;
  if (tile >= ngroups * mtiles) return;
  int tg = tile % ngroups;
  int tm = tile / ngroups;
  int hi = lane >> 4;
  int lo = lane & 15;

  int arow = tm * 16 + lo;
  float s = 1.f, t = 0.f;
  if (useBN) {
    int node = arow % NNODE;
    float rs = rstd[node];
    s = gamma[node] * rs;
    t = beta[node] - mean[node] * s;
  }
  const float* ap = in + (long)arow * K;
  const __bf16* bp0 = Wb + (long)(tg * 64 +  0 + lo) * K + hi * 16;
  const __bf16* bp1 = Wb + (long)(tg * 64 + 16 + lo) * K + hi * 16;
  const __bf16* bp2 = Wb + (long)(tg * 64 + 32 + lo) * K + hi * 16;
  const __bf16* bp3 = Wb + (long)(tg * 64 + 48 + lo) * K + hi * 16;

  v8f acc0 = {0.f,0.f,0.f,0.f,0.f,0.f,0.f,0.f};
  v8f acc1 = acc0, acc2 = acc0, acc3 = acc0;

  for (int k0 = 0; k0 < K; k0 += 32) {
    // A fragment: contiguous K (k = 16*hi + s), BN fused.
    v16bf a;
#pragma unroll
    for (int p = 0; p < 8; ++p) {
      int ka = k0 + 2 * p + hi * 16;
      v2f f = *(const v2f*)(ap + ka);
      a[2 * p]     = (__bf16)(f.x * s + t);
      a[2 * p + 1] = (__bf16)(f.y * s + t);
    }
    // B fragments: 16 contiguous bf16 each (32B vector load).
    v16bf b0 = *(const v16bf*)(bp0 + k0);
    v16bf b1 = *(const v16bf*)(bp1 + k0);
    v16bf b2 = *(const v16bf*)(bp2 + k0);
    v16bf b3 = *(const v16bf*)(bp3 + k0);
    acc0 = __builtin_amdgcn_wmma_f32_16x16x32_bf16(false, a, false, b0, (short)0, acc0, false, false);
    acc1 = __builtin_amdgcn_wmma_f32_16x16x32_bf16(false, a, false, b1, (short)0, acc1, false, false);
    acc2 = __builtin_amdgcn_wmma_f32_16x16x32_bf16(false, a, false, b2, (short)0, acc2, false, false);
    acc3 = __builtin_amdgcn_wmma_f32_16x16x32_bf16(false, a, false, b3, (short)0, acc3, false, false);
  }

  // Epilogue: C layout row = r + hi*8, col = lane lo within each 16-col tile.
#pragma unroll
  for (int nt = 0; nt < 4; ++nt) {
    v8f acc = (nt == 0) ? acc0 : (nt == 1) ? acc1 : (nt == 2) ? acc2 : acc3;
    int ncol = tg * 64 + nt * 16 + lo;
    float bb = bias[ncol];
#pragma unroll
    for (int r = 0; r < 8; ++r) {
      int orow = tm * 16 + r + hi * 8;
      float v = acc[r] + bb;
      if (act == 1)      v = (v >= 0.f) ? v : 0.2f * v;
      else if (act == 2) v = (v > 0.f)  ? v : 0.f;
      out[(long)orow * Nout + ncol] = v;
    }
  }
}

// ---------------------------------------------------------------------------
// Entry layer (K=2): out = lrelu( bn(x) @ W^T + b ), W: [64,2]. Trivial FLOPs.
// ---------------------------------------------------------------------------
__global__ void msg_entry_kernel(const float* __restrict__ x,
                                 const float* __restrict__ W,
                                 const float* __restrict__ bias,
                                 const float* __restrict__ mean,
                                 const float* __restrict__ rstd,
                                 const float* __restrict__ gamma,
                                 const float* __restrict__ beta,
                                 float* __restrict__ out) {
  int idx = blockIdx.x * blockDim.x + threadIdx.x;   // ROWS*64
  if (idx >= ROWS * 64) return;
  int row = idx >> 6, o = idx & 63;
  int node = row % NNODE;
  float s = gamma[node] * rstd[node];
  float t = beta[node] - mean[node] * s;
  float x0 = x[(long)row * 2]     * s + t;
  float x1 = x[(long)row * 2 + 1] * s + t;
  float v = x0 * W[o * 2] + x1 * W[o * 2 + 1] + bias[o];
  out[(long)row * 64 + o] = (v >= 0.f) ? v : 0.2f * v;
}

// ---------------------------------------------------------------------------
// Exit layer (Nout=2): out = bn(in) @ W^T + b, W: [2,128]. Trivial FLOPs.
// ---------------------------------------------------------------------------
__global__ void out_exit_kernel(const float* __restrict__ in,
                                const float* __restrict__ W,
                                const float* __restrict__ bias,
                                const float* __restrict__ mean,
                                const float* __restrict__ rstd,
                                const float* __restrict__ gamma,
                                const float* __restrict__ beta,
                                float* __restrict__ out) {
  int idx = blockIdx.x * blockDim.x + threadIdx.x;   // ROWS*2
  if (idx >= ROWS * 2) return;
  int row = idx >> 1, o = idx & 1;
  int node = row % NNODE;
  float s = gamma[node] * rstd[node];
  float t = beta[node] - mean[node] * s;
  const float* ip = in + (long)row * 128;
  const float* wp = W + o * 128;
  float sum = bias[o];
#pragma unroll 4
  for (int k = 0; k < 128; ++k) sum += (ip[k] * s + t) * wp[k];
  out[(long)row * 2 + o] = sum;
}

// ---------------------------------------------------------------------------
// m = mean_over_nodes(h) - h      (h: [B,23,F])
// ---------------------------------------------------------------------------
__global__ void mean_sub_kernel(const float* __restrict__ in,
                                float* __restrict__ out, int F) {
  int idx = blockIdx.x * blockDim.x + threadIdx.x;   // B*F
  if (idx >= BATCH * F) return;
  int b = idx / F, f = idx % F;
  float sum = 0.f;
  for (int n = 0; n < NNODE; ++n) sum += in[((long)b * NNODE + n) * F + f];
  float mn = sum / (float)NNODE;
  for (int n = 0; n < NNODE; ++n) {
    long o = ((long)b * NNODE + n) * F + f;
    out[o] = mn - in[o];
  }
}

// ---------------------------------------------------------------------------
// out[b,n,:] = concat(att[b,n,:], mean_over_nodes(att)[b,:])  (128 -> 256)
// ---------------------------------------------------------------------------
__global__ void mean_cat_kernel(const float* __restrict__ att,
                                float* __restrict__ out) {
  int idx = blockIdx.x * blockDim.x + threadIdx.x;   // B*128
  if (idx >= BATCH * 128) return;
  int b = idx / 128, f = idx % 128;
  float sum = 0.f;
  for (int n = 0; n < NNODE; ++n) sum += att[((long)b * NNODE + n) * 128 + f];
  float mn = sum / (float)NNODE;
  for (int n = 0; n < NNODE; ++n) {
    float v = att[((long)b * NNODE + n) * 128 + f];
    long o = ((long)b * NNODE + n) * 256;
    out[o + f]       = v;
    out[o + 128 + f] = mn;
  }
}

// ---------------------------------------------------------------------------
// Flash attention across the batch axis.
// qp: [B,23,128] f32 (cols h*32+d); kb: same layout in bf16;
// vt: [n][h][d][m] bf16 (m contiguous).  One wave owns (n, h, 16-row l-tile).
// Per 32-wide m-chunk: S = (Q/sqrt(hd)) K^T (2 WMMAs), online softmax,
// P staged through per-wave LDS (C-layout -> A-layout), acc += P V (2 WMMAs).
// ---------------------------------------------------------------------------
__global__ void attn_kernel(const float* __restrict__ qp,
                            const __bf16* __restrict__ kb,
                            const __bf16* __restrict__ vt,
                            float* __restrict__ att) {
  __shared__ __bf16 pstage[4][16 * 32];
  int lane = threadIdx.x & 31;
  int wave = threadIdx.x >> 5;
  int wid = blockIdx.x * 4 + wave;
  if (wid >= NNODE * 4 * (BATCH / 16)) return;
  int l0 = (wid % (BATCH / 16)) * 16;
  int nh = wid / (BATCH / 16);
  int h = nh & 3, n = nh >> 2;
  int hi = lane >> 4, lo = lane & 15;
  const float scale = 0.17677669529663687f;   // 1/sqrt(32)

  // Q A-fragment: contiguous d (d = 16*hi + s), scale fused.
  v16bf aq;
  {
    const float* qr = qp + ((long)(l0 + lo) * NNODE + n) * 128 + h * 32;
#pragma unroll
    for (int p = 0; p < 8; ++p) {
      int d = 2 * p + hi * 16;
      v2f f = *(const v2f*)(qr + d);
      aq[2 * p]     = (__bf16)(f.x * scale);
      aq[2 * p + 1] = (__bf16)(f.y * scale);
    }
  }

  // Per-lane base pointers (advance by m inside the loop).
  const __bf16* krow0 = kb + ((long)lo * NNODE + n) * 128 + h * 32 + hi * 16;
  const __bf16* krow1 = kb + ((long)(16 + lo) * NNODE + n) * 128 + h * 32 + hi * 16;
  const long  kstep  = (long)NNODE * 128;          // one m step in kb
  const __bf16* vrow0 = vt + ((long)((n * 4 + h) * 32 + lo)      * BATCH) + hi * 16;
  const __bf16* vrow1 = vt + ((long)((n * 4 + h) * 32 + 16 + lo) * BATCH) + hi * 16;

  float mrun[8], lrun[8];
#pragma unroll
  for (int r = 0; r < 8; ++r) { mrun[r] = -1e30f; lrun[r] = 0.f; }
  v8f acc0 = {0.f,0.f,0.f,0.f,0.f,0.f,0.f,0.f};
  v8f acc1 = acc0;
  __bf16* pst = &pstage[wave][0];

  for (int m0 = 0; m0 < BATCH; m0 += 32) {
    // K^T B-fragments: single 32B vector load each.
    v16bf bk0 = *(const v16bf*)(krow0 + (long)m0 * kstep);
    v16bf bk1 = *(const v16bf*)(krow1 + (long)m0 * kstep);
    v8f z = {0.f,0.f,0.f,0.f,0.f,0.f,0.f,0.f};
    v8f s0 = __builtin_amdgcn_wmma_f32_16x16x32_bf16(false, aq, false, bk0, (short)0, z, false, false);
    v8f s1 = __builtin_amdgcn_wmma_f32_16x16x32_bf16(false, aq, false, bk1, (short)0, z, false, false);

    // Online softmax per owned row r (C layout: row = r + hi*8).
#pragma unroll
    for (int r = 0; r < 8; ++r) {
      float cm = fmaxf(s0[r], s1[r]);
      cm = fmaxf(cm, __shfl_xor(cm, 1, 32));
      cm = fmaxf(cm, __shfl_xor(cm, 2, 32));
      cm = fmaxf(cm, __shfl_xor(cm, 4, 32));
      cm = fmaxf(cm, __shfl_xor(cm, 8, 32));
      float nm   = fmaxf(mrun[r], cm);
      float corr = __expf(mrun[r] - nm);
      float p0   = __expf(s0[r] - nm);
      float p1   = __expf(s1[r] - nm);
      float rs = p0 + p1;
      rs += __shfl_xor(rs, 1, 32);
      rs += __shfl_xor(rs, 2, 32);
      rs += __shfl_xor(rs, 4, 32);
      rs += __shfl_xor(rs, 8, 32);
      lrun[r] = lrun[r] * corr + rs;
      mrun[r] = nm;
      acc0[r] *= corr;
      acc1[r] *= corr;
      int prow = r + hi * 8;
      pst[prow * 32 + lo]      = (__bf16)p0;
      pst[prow * 32 + 16 + lo] = (__bf16)p1;
    }
    asm volatile("s_wait_dscnt 0x0" ::: "memory");  // intra-wave LDS RAW
    // Reload P as A-fragment (row = lo, K over m-chunk).
    v16bf ap;
#pragma unroll
    for (int i = 0; i < 16; ++i) {
      int kk = i + hi * 16;
      ap[i] = pst[lo * 32 + kk];
    }
    // V B-fragments: contiguous along m in vt -> single 32B load each.
    v16bf bv0 = *(const v16bf*)(vrow0 + m0);
    v16bf bv1 = *(const v16bf*)(vrow1 + m0);
    acc0 = __builtin_amdgcn_wmma_f32_16x16x32_bf16(false, ap, false, bv0, (short)0, acc0, false, false);
    acc1 = __builtin_amdgcn_wmma_f32_16x16x32_bf16(false, ap, false, bv1, (short)0, acc1, false, false);
  }

#pragma unroll
  for (int r = 0; r < 8; ++r) {
    int b = l0 + r + hi * 8;
    float inv = 1.f / lrun[r];
    float* orow = att + ((long)b * NNODE + n) * 128 + h * 32;
    orow[lo]      = acc0[r] * inv;
    orow[16 + lo] = acc1[r] * inv;
  }
}

// ---------------------------------------------------------------------------
// Host-side orchestration.  Input index map (JAX pytree sorted-key order):
//  0 attn.Win[384,128] 1 attn.bin 2 attn.Wout[128,128] 3 attn.bout
//  4..7 k[0](g,b,W,bias)  8..11 k[1]
// 12..15 msg[0] (W[64,2]) 16..19 msg[1] (W[128,64]) 20..23 msg[2] (W[128,128])
// 24..25 outp1[0](W[256,256],b) 26..27 outp1[1](W[128,256],b)
// 28..29 outp2(W[2,128],b) 30..31 outp2_bn(g,b)
// 32..39 q[0],q[1]  40..47 v[0],v[1]  48 x[1024,23,2]
// ---------------------------------------------------------------------------
extern "C" void kernel_launch(void* const* d_in, const int* in_sizes, int n_in,
                              void* d_out, int out_size, void* d_ws, size_t ws_size,
                              hipStream_t stream) {
  (void)in_sizes; (void)n_in; (void)out_size; (void)ws_size;
  const long R = ROWS;
  float* ws     = (float*)d_ws;
  float* buf64  = ws;                    // R*64
  float* bufA   = buf64 + R * 64;        // R*128 each
  float* bufB   = bufA  + R * 128;
  float* bufC   = bufB  + R * 128;
  float* bufD   = bufC  + R * 128;
  float* bufE   = bufD  + R * 128;
  float* bufCat = bufE  + R * 128;       // R*256
  float* bufO1  = bufCat + R * 256;      // R*256
  float* st     = bufO1 + R * 256;       // 8 stat sets, 64 floats each
  __bf16* wb    = (__bf16*)(st + 512);   // bf16 weights, 286720 elems
  __bf16* kb    = wb + 286720;           // bf16 K, R*128
  __bf16* vt    = kb + R * 128;          // bf16 V^T, 23*4*32*1024
  auto stM = [&](int j) { return st + j * 64; };
  auto stR = [&](int j) { return st + j * 64 + 32; };

  const float* IN[49];
  for (int i = 0; i < 49; ++i) IN[i] = (const float*)d_in[i];
  const float* x = IN[48];

  // bf16 weight region offsets
  __bf16* wb_msg1 = wb;                   // 128*64
  __bf16* wb_msg2 = wb_msg1 + 8192;       // 128*128
  __bf16* wb_q0   = wb_msg2 + 16384;
  __bf16* wb_q1   = wb_q0   + 16384;
  __bf16* wb_k0   = wb_q1   + 16384;
  __bf16* wb_k1   = wb_k0   + 16384;
  __bf16* wb_v0   = wb_k1   + 16384;
  __bf16* wb_v1   = wb_v0   + 16384;
  __bf16* wb_win  = wb_v1   + 16384;      // 384*128
  __bf16* wb_wout = wb_win  + 49152;      // 128*128
  __bf16* wb_o10  = wb_wout + 16384;      // 256*256
  __bf16* wb_o11  = wb_o10  + 65536;      // 128*256

  auto cvt = [&](const float* src, __bf16* dst, int nelem) {
    cvt_bf16_kernel<<<(nelem + 255) / 256, 256, 0, stream>>>(src, dst, nelem);
  };
  cvt(IN[18], wb_msg1, 8192);
  cvt(IN[22], wb_msg2, 16384);
  cvt(IN[34], wb_q0, 16384);  cvt(IN[38], wb_q1, 16384);
  cvt(IN[6],  wb_k0, 16384);  cvt(IN[10], wb_k1, 16384);
  cvt(IN[42], wb_v0, 16384);  cvt(IN[46], wb_v1, 16384);
  cvt(IN[0],  wb_win, 49152);
  cvt(IN[2],  wb_wout, 16384);
  cvt(IN[24], wb_o10, 65536);
  cvt(IN[26], wb_o11, 32768);

  auto gemm = [&](const float* in, const __bf16* W, const float* bias,
                  const float* mean, const float* rstd,
                  const float* g, const float* be,
                  float* out, int K, int Nout, int act, int useBN) {
    int ngroups = Nout / 64;
    int tiles = (int)(R / 16) * ngroups;
    int blocks = (tiles + 3) / 4;
    gemm_bn_wmma_kernel<<<blocks, 128, 0, stream>>>(in, W, bias, mean, rstd, g, be,
                                                    out, (int)R, K, Nout, act, useBN);
  };
  auto stats = [&](const float* in, int F, int j) {
    bn_stats_kernel<<<NNODE, 256, 0, stream>>>(in, F, stM(j), stR(j));
  };

  // message MLP: (BN -> Lin -> LReLU) x2, BN -> Lin
  stats(x, 2, 0);
  msg_entry_kernel<<<(int)(R * 64 + 255) / 256, 256, 0, stream>>>(
      x, IN[14], IN[15], stM(0), stR(0), IN[12], IN[13], buf64);
  stats(buf64, 64, 1);
  gemm(buf64, wb_msg1, IN[19], stM(1), stR(1), IN[16], IN[17], bufA, 64, 128, 1, 1);
  stats(bufA, 128, 2);
  gemm(bufA, wb_msg2, IN[23], stM(2), stR(2), IN[20], IN[21], bufB, 128, 128, 0, 1);

  // m = mean_nodes - h   -> bufA
  mean_sub_kernel<<<(BATCH * 128 + 255) / 256, 256, 0, stream>>>(bufB, bufA, 128);
  stats(bufA, 128, 3);    // shared BN stats for q/k/v first blocks

  // q branch -> bufC
  gemm(bufA, wb_q0, IN[35], stM(3), stR(3), IN[32], IN[33], bufB, 128, 128, 1, 1);
  stats(bufB, 128, 4);
  gemm(bufB, wb_q1, IN[39], stM(4), stR(4), IN[36], IN[37], bufC, 128, 128, 1, 1);
  // k branch -> bufD
  gemm(bufA, wb_k0, IN[7], stM(3), stR(3), IN[4], IN[5], bufB, 128, 128, 1, 1);
  stats(bufB, 128, 5);
  gemm(bufB, wb_k1, IN[11], stM(5), stR(5), IN[8], IN[9], bufD, 128, 128, 1, 1);
  // v branch -> bufE
  gemm(bufA, wb_v0, IN[43], stM(3), stR(3), IN[40], IN[41], bufB, 128, 128, 1, 1);
  stats(bufB, 128, 6);
  gemm(bufB, wb_v1, IN[47], stM(6), stR(6), IN[44], IN[45], bufE, 128, 128, 1, 1);

  // attention in-projections: qp->bufA, kp->bufB, vp->bufC
  gemm(bufC, wb_win,             IN[1],       nullptr, nullptr, nullptr, nullptr,
       bufA, 128, 128, 0, 0);
  gemm(bufD, wb_win + 128 * 128, IN[1] + 128, nullptr, nullptr, nullptr, nullptr,
       bufB, 128, 128, 0, 0);
  gemm(bufE, wb_win + 256 * 128, IN[1] + 256, nullptr, nullptr, nullptr, nullptr,
       bufC, 128, 128, 0, 0);

  // bf16 staging for attention operands
  cvt(bufB, kb, (int)(R * 128));
  v_pack_kernel<<<(NNODE * 4 * 32 * BATCH + 255) / 256, 256, 0, stream>>>(bufC, vt);

  // flash attention across batch axis -> bufD
  attn_kernel<<<(NNODE * 4 * (BATCH / 16)) / 4, 128, 0, stream>>>(bufA, kb, vt, bufD);

  // output projection -> bufE
  gemm(bufD, wb_wout, IN[3], nullptr, nullptr, nullptr, nullptr, bufE, 128, 128, 0, 0);

  // concat(att, mean_nodes(att)) -> bufCat [R,256]
  mean_cat_kernel<<<(BATCH * 128 + 255) / 256, 256, 0, stream>>>(bufE, bufCat);

  // outp1: Lin 256->256 ReLU, Lin 256->128 ReLU
  gemm(bufCat, wb_o10, IN[25], nullptr, nullptr, nullptr, nullptr, bufO1, 256, 256, 2, 0);
  gemm(bufO1,  wb_o11, IN[27], nullptr, nullptr, nullptr, nullptr, bufA, 256, 128, 2, 0);

  // final: BN -> Lin 128->2
  stats(bufA, 128, 7);
  out_exit_kernel<<<(int)(R * 2 + 255) / 256, 256, 0, stream>>>(
      bufA, IN[28], IN[29], stM(7), stR(7), IN[30], IN[31], (float*)d_out);
}
